// Conv2dBN_qat_int8_39384850104587
// MI455X (gfx1250) — 8-bit matrix operand order repaired and checked by reading, compiled, not run on hardware
//
#include <hip/hip_runtime.h>
#include <math.h>


typedef __attribute__((ext_vector_type(8))) int v8i;

#define B_    4
#define C_    32
#define H_    28
#define W_    28
#define O_    64
#define KH    3
#define KW    3
#define KDIM  (C_ * KH * KW)
#define KBLK  5
#define MDIM  (B_ * H_ * W_)
#define MT    (MDIM / 16)
#define NT    (O_ / 16)
#define EPSV  1e-5f
#define MOMV  0.1f

__global__ void pack_a_kernel(const float* __restrict__ x,
                              const float* __restrict__ sf_ptr,
                              int use_safe,
                              unsigned int* __restrict__ Apk)
{
    float s = sf_ptr[0];
    if (use_safe) s = fabsf(s) + 1e-8f;
    const float inv = 1.0f / s;

    const int total = MT * KBLK * 256;
    for (int idx = blockIdx.x * blockDim.x + threadIdx.x; idx < total;
         idx += gridDim.x * blockDim.x) {
        const int tile   = idx >> 8;
        const int within = idx & 255;
        const int lane   = within >> 3;
        const int v      = within & 7;
        const int tm     = tile / KBLK;
        const int kb     = tile % KBLK;
        const int hi     = lane >> 4;

        const int m   = tm * 16 + (lane & 15);
        const int b   = m / (H_ * W_);
        const int pix = m % (H_ * W_);
        const int oh  = pix / W_;
        const int ow  = pix % W_;

        unsigned int dword = 0;
#pragma unroll
        for (int by = 0; by < 4; ++by) {
            const int k = kb * 64 + (v >> 1) * 16 + hi * 8 + (v & 1) * 4 + by;
            int q = 0;
            if (k < KDIM) {
                const int c  = k / (KH * KW);
                const int kk = k % (KH * KW);
                const int kh = kk / KW;
                const int kw = kk % KW;
                const int ih = oh + kh - 1;
                const int iw = ow + kw - 1;
                if (ih >= 0 && ih < H_ && iw >= 0 && iw < W_) {
                    const float val = x[((b * C_ + c) * H_ + ih) * W_ + iw];
                    float qf = rintf(val * inv);
                    qf = fminf(fmaxf(qf, -128.0f), 127.0f);
                    q = (int)qf;
                }
            }
            dword |= ((unsigned int)(q & 0xFF)) << (8 * by);
        }
        Apk[idx] = dword;
    }
}

__global__ void pack_w_kernel(const float* __restrict__ w,
                              const float* __restrict__ sw,
                              const float* __restrict__ wf,
                              int use_safe,
                              unsigned int* __restrict__ Bpk)
{
    const int total = NT * KBLK * 256;
    for (int idx = blockIdx.x * blockDim.x + threadIdx.x; idx < total;
         idx += gridDim.x * blockDim.x) {
        const int tile   = idx >> 8;
        const int within = idx & 255;
        const int lane   = within >> 3;
        const int v      = within & 7;
        const int tn     = tile / KBLK;
        const int kb     = tile % KBLK;
        const int hi     = lane >> 4;

        const int o = tn * 16 + (lane & 15);
        const float f = wf ? wf[o] : 1.0f;
        float s = sw[o] * f;
        if (use_safe) s = fabsf(s) + 1e-8f;
        const float inv = 1.0f / s;

        unsigned int dword = 0;
#pragma unroll
        for (int by = 0; by < 4; ++by) {
            const int k = kb * 64 + (v >> 1) * 16 + hi * 8 + (v & 1) * 4 + by;
            int q = 0;
            if (k < KDIM) {
                const float val = w[o * KDIM + k] * f;
                float qf = rintf(val * inv);
                qf = fminf(fmaxf(qf, -128.0f), 127.0f);
                q = (int)qf;
            }
            dword |= ((unsigned int)(q & 0xFF)) << (8 * by);
        }
        Bpk[idx] = dword;
    }
}

__global__ void gemm_iu8_kernel(const unsigned int* __restrict__ Apk,
                                const unsigned int* __restrict__ Bpk,
                                int* __restrict__ acc_out)
{
    const int lane = threadIdx.x & 31;
    const int wave = blockIdx.x * (blockDim.x >> 5) + (threadIdx.x >> 5);
    if (wave >= MT * NT) return;
    const int tm = wave / NT;
    const int tn = wave % NT;

    v8i acc = {};
#pragma unroll
    for (int kb = 0; kb < KBLK; ++kb) {
        const unsigned int* ap = Apk + (tm * KBLK + kb) * 256 + lane * 8;
        const unsigned int* bp = Bpk + (tn * KBLK + kb) * 256 + lane * 8;
        if (kb + 1 < KBLK) {
            __builtin_prefetch(ap + 256, 0, 1);
            __builtin_prefetch(bp + 256, 0, 1);
        }
        v8i a = *(const v8i*)ap;
        v8i b = *(const v8i*)bp;
        acc = __builtin_amdgcn_wmma_i32_16x16x64_iu8(true, a, true, b, acc,
                                                     false, false);
    }

    const int n     = tn * 16 + (lane & 15);
    const int mbase = tm * 16 + (lane >> 4) * 8;
#pragma unroll
    for (int r = 0; r < 8; ++r)
        acc_out[(mbase + r) * O_ + n] = acc[r];
}

__global__ void stats_kernel(const int* __restrict__ acc,
                             const float* __restrict__ sf_ptr,
                             const float* __restrict__ sw,
                             const float* __restrict__ gamma,
                             const float* __restrict__ beta,
                             const float* __restrict__ running_var,
                             float* __restrict__ par)
{
    __shared__ float ssum[256];
    __shared__ float ssq[256];
    const int o = blockIdx.x;
    const float scale = sf_ptr[0] * sw[o];

    float sum = 0.0f, sq = 0.0f;
    for (int m = threadIdx.x; m < MDIM; m += blockDim.x) {
        const float v = (float)acc[m * O_ + o] * scale;
        sum += v;
        sq  += v * v;
    }
    ssum[threadIdx.x] = sum;
    ssq[threadIdx.x]  = sq;
    __syncthreads();
    for (int st = 128; st > 0; st >>= 1) {
        if ((int)threadIdx.x < st) {
            ssum[threadIdx.x] += ssum[threadIdx.x + st];
            ssq[threadIdx.x]  += ssq[threadIdx.x + st];
        }
        __syncthreads();
    }
    if (threadIdx.x == 0) {
        const float bm = ssum[0] / (float)MDIM;
        float bv = ssq[0] / (float)MDIM - bm * bm;
        bv = fmaxf(bv, 0.0f);
        const float bstd   = sqrtf(bv + EPSV);
        const float rv_new = MOMV * bv + (1.0f - MOMV) * running_var[o];
        const float wfv    = gamma[o] / sqrtf(rv_new + EPSV);
        par[o]        = wfv;
        par[64 + o]   = fabsf(sw[o] * wfv) + 1e-8f;
        par[128 + o]  = beta[o] - gamma[o] * bm / bstd;
        par[192 + o]  = sqrtf(rv_new + EPSV) / bstd;
    }
}

__global__ void finalize_kernel(const int* __restrict__ acc,
                                const float* __restrict__ sf_ptr,
                                const float* __restrict__ so_ptr,
                                const float* __restrict__ par,
                                float* __restrict__ out)
{
    const float sf_safe = fabsf(sf_ptr[0]) + 1e-8f;
    const float so      = so_ptr[0];
    const float inv_so  = 1.0f / so;

    const int total = B_ * O_ * H_ * W_;
    for (int idx = blockIdx.x * blockDim.x + threadIdx.x; idx < total;
         idx += gridDim.x * blockDim.x) {
        const int b   = idx / (O_ * H_ * W_);
        const int rem = idx % (O_ * H_ * W_);
        const int o   = rem / (H_ * W_);
        const int pix = rem % (H_ * W_);
        const int m   = b * (H_ * W_) + pix;

        float v = (float)acc[m * O_ + o] * (sf_safe * par[64 + o]);
        v = v * par[192 + o] + par[128 + o];
        float q = fminf(fmaxf(rintf(v * inv_so), -128.0f), 127.0f);
        out[idx] = q * so;
    }
}

extern "C" void kernel_launch(void* const* d_in, const int* in_sizes, int n_in,
                              void* d_out, int out_size, void* d_ws, size_t ws_size,
                              hipStream_t stream)
{
    (void)in_sizes; (void)n_in; (void)out_size; (void)ws_size;

    const float* x      = (const float*)d_in[0];
    const float* weight = (const float*)d_in[1];
    const float* sf     = (const float*)d_in[3];
    const float* sw     = (const float*)d_in[4];
    const float* so     = (const float*)d_in[5];
    const float* gamma  = (const float*)d_in[6];
    const float* beta   = (const float*)d_in[7];
    const float* rvar   = (const float*)d_in[9];
    float*       out    = (float*)d_out;

    char* ws = (char*)d_ws;
    unsigned int* Apk = (unsigned int*)(ws);
    unsigned int* Bpk = (unsigned int*)(ws + 1003520);
    int*          acc = (int*)(ws + 1024000);
    float*        par = (float*)(ws + 1826816);

    const int PACK_A_BLK = (MT * KBLK * 256 + 255) / 256;
    const int PACK_W_BLK = (NT * KBLK * 256 + 255) / 256;
    const int GEMM_BLK   = (MT * NT) / 8;
    const int FIN_BLK    = (B_ * O_ * H_ * W_ + 255) / 256;

    pack_a_kernel<<<PACK_A_BLK, 256, 0, stream>>>(x, sf, 0, Apk);
    pack_w_kernel<<<PACK_W_BLK, 256, 0, stream>>>(weight, sw, nullptr, 0, Bpk);
    gemm_iu8_kernel<<<GEMM_BLK, 256, 0, stream>>>(Apk, Bpk, acc);
    stats_kernel<<<O_, 256, 0, stream>>>(acc, sf, sw, gamma, beta, rvar, par);

    pack_a_kernel<<<PACK_A_BLK, 256, 0, stream>>>(x, sf, 1, Apk);
    pack_w_kernel<<<PACK_W_BLK, 256, 0, stream>>>(weight, sw, par  , 1, Bpk);
    gemm_iu8_kernel<<<GEMM_BLK, 256, 0, stream>>>(Apk, Bpk, acc);

    finalize_kernel<<<FIN_BLK, 256, 0, stream>>>(acc, sf, so, par, out);
}
